// feature_extractor_84902913507990
// MI455X (gfx1250) — matrix operand order repaired and checked by reading, not compiled, not run on hardware
//
#include <hip/hip_runtime.h>
#include <stdint.h>

typedef __attribute__((ext_vector_type(16))) _Float16 v16h;
typedef __attribute__((ext_vector_type(8)))  float    v8f;

union Frag16 { v16h v; uint4 q[2]; _Float16 h[16]; };

#define BATCH 4
#define NPTS  8192
#define KNN   20
#define INV_BN 0.9999950000374997f   // 1/sqrt(1+1e-5)

// ---- d_out layout (float elements, outputs concatenated in return order) ----
#define OFF_FEAT    ((size_t)0)                              // B*512*N
#define OFF_IDXFLAT ((size_t)BATCH * 512 * NPTS)             // B*N*K
#define OFF_KNN     (OFF_IDXFLAT + (size_t)BATCH*NPTS*KNN)   // B*N*K*3
#define OFF_IDX     (OFF_KNN + (size_t)BATCH*NPTS*KNN*3)     // B*N*K

// ---- workspace layout (bytes) ----
#define WS_IDX 0                                   // int32 [B*N*K]
#define WS_WH  ((size_t)BATCH*NPTS*KNN*4)          // f16 weights, 143360 halfs
#define WS_CAT (WS_WH + 286720)                    // f16 [B][N][256]
// half-offsets inside WS_WH
#define WH_W1 0        // 32 x 32 (K padded 3->32)
#define WH_W2 1024     // 32 x 32
#define WH_W3 2048     // 64 x 32
#define WH_W4 4096     // 128 x 64
#define WH_W5 12288    // 512 x 256
#define WH_TOTAL 143360

__device__ __forceinline__ v8f wmma16(const Frag16& a, const Frag16& b, v8f c) {
  return __builtin_amdgcn_wmma_f32_16x16x32_f16(false, a.v, false, b.v,
                                                (short)0, c, false, false);
}

// A fragment (16x32 f16): lane ln = row M, lane-half lg; K chunks [kblk+lg*8, +8) and [kblk+16+lg*8, +8)
__device__ __forceinline__ Frag16 load_a(const _Float16* __restrict__ w, int kpad,
                                         int mbase, int ln, int lg, int kblk) {
  Frag16 f;
  const _Float16* p = w + (size_t)(mbase + ln) * kpad + kblk + lg * 8;
  f.q[0] = *(const uint4*)(p);
  f.q[1] = *(const uint4*)(p + 16);
  return f;
}

// B fragment (32x16 f16) from LDS activation buffer hbuf[col*64 + k]:
// lane ln = column N, lane-half lg; K chunks [kblk+lg*8, +8) and [kblk+16+lg*8, +8)
__device__ __forceinline__ Frag16 load_b_lds(const _Float16* hb, int colbase,
                                             int ln, int lg, int kblk) {
  Frag16 f;
  const _Float16* p = hb + (size_t)(colbase + ln) * 64 + kblk + lg * 8;
  f.q[0] = *(const uint4*)(p);
  f.q[1] = *(const uint4*)(p + 16);
  return f;
}

// ---------------- Kernel 1: kNN top-20 ----------------
__global__ __launch_bounds__(256) void knn_kernel(const float* __restrict__ x,
                                                  float* __restrict__ out,
                                                  int* __restrict__ ws_idx) {
  __shared__ __align__(16) float xl[3 * 2048];
  int b     = blockIdx.x >> 5;            // 32 blocks per batch
  int pbase = (blockIdx.x & 31) * 256;
  int i     = pbase + threadIdx.x;
  const float* xb = x + (size_t)b * 3 * NPTS;
  float xi0 = xb[i], xi1 = xb[NPTS + i], xi2 = xb[2 * NPTS + i];

  float bd[KNN]; int bi[KNN];
#pragma unroll
  for (int t = 0; t < KNN; ++t) { bd[t] = 3.4e38f; bi[t] = 0x7fffffff; }

  for (int chunk = 0; chunk < 4; ++chunk) {
    int jb = chunk * 2048;
    __syncthreads();
#pragma unroll
    for (int c = 0; c < 3; ++c)
      for (int t = threadIdx.x; t < 512; t += 256)
        ((float4*)(xl + c * 2048))[t] = ((const float4*)(xb + c * NPTS + jb))[t];
    __syncthreads();
    for (int jj = 0; jj < 2048; ++jj) {
      int j = jb + jj;
      float dx = xl[jj] - xi0, dy = xl[2048 + jj] - xi1, dz = xl[4096 + jj] - xi2;
      float d2 = dx * dx + dy * dy + dz * dz;
      // ordering key (d2 asc, j asc) == top_k of -d2 with stable ties
      if (d2 < bd[KNN - 1] || (d2 == bd[KNN - 1] && j < bi[KNN - 1])) {
#pragma unroll
        for (int t = KNN - 1; t > 0; --t) {
          bool lt_t  = (d2 < bd[t])     || (d2 == bd[t]     && j < bi[t]);
          bool lt_tm = (d2 < bd[t - 1]) || (d2 == bd[t - 1] && j < bi[t - 1]);
          if (lt_t) {
            if (lt_tm) { bd[t] = bd[t - 1]; bi[t] = bi[t - 1]; }
            else       { bd[t] = d2;        bi[t] = j; }
          }
        }
        if (d2 < bd[0] || (d2 == bd[0] && j < bi[0])) { bd[0] = d2; bi[0] = j; }
      }
    }
  }

  size_t base = ((size_t)b * NPTS + i) * KNN;
#pragma unroll
  for (int kk = 0; kk < KNN; ++kk) {
    int j = bi[kk];
    ws_idx[base + kk]             = j;
    out[OFF_IDX + base + kk]      = (float)j;
    out[OFF_IDXFLAT + base + kk]  = (float)(b * NPTS + j);
    out[OFF_KNN + (base + kk) * 3 + 0] = xb[j];
    out[OFF_KNN + (base + kk) * 3 + 1] = xb[NPTS + j];
    out[OFF_KNN + (base + kk) * 3 + 2] = xb[2 * NPTS + j];
  }
}

// ---------------- Kernel 2: convert weights to f16 ----------------
__global__ __launch_bounds__(256) void wconv_kernel(
    const float* __restrict__ W1, const float* __restrict__ W2,
    const float* __restrict__ W3, const float* __restrict__ W4,
    const float* __restrict__ W5, _Float16* __restrict__ wh) {
  int t = blockIdx.x * 256 + threadIdx.x;
  if (t >= WH_TOTAL) return;
  float v;
  if (t < WH_W2)      { int r = t >> 5, c = t & 31; v = (c < 3) ? W1[r * 3 + c] : 0.f; }
  else if (t < WH_W3) { v = W2[t - WH_W2]; }
  else if (t < WH_W4) { v = W3[t - WH_W3]; }
  else if (t < WH_W5) { v = W4[t - WH_W4]; }
  else                { v = W5[t - WH_W5]; }
  wh[t] = (_Float16)v;
}

// ---------------- Kernel 3: fused EdgeConv layers 1..4 + max-pool ----------------
__global__ __launch_bounds__(128) void edgeconv_kernel(
    const float* __restrict__ x, const int* __restrict__ ws_idx,
    const _Float16* __restrict__ wh,
    const float* __restrict__ g1, const float* __restrict__ b1,
    const float* __restrict__ g2, const float* __restrict__ b2,
    const float* __restrict__ g3, const float* __restrict__ b3,
    const float* __restrict__ g4, const float* __restrict__ b4,
    _Float16* __restrict__ cat) {
  __shared__ __align__(16) _Float16 hbuf[320 * 64];  // 40KB: [col][ch]
  __shared__ int smax[256 * 16];                     // 16KB: [cat_ch][point]
  int b     = blockIdx.x >> 9;                       // 512 tiles per batch
  int pbase = (blockIdx.x & 511) * 16;
  int tid   = threadIdx.x;
  const float* xb = x + (size_t)b * 3 * NPTS;

  // zero K rows 0..31 of every column (layer-1 K padding), and the max buffer
  for (int t = tid; t < 320 * 16; t += 128) {
    int col = t >> 4, q = t & 15;
    ((uint32_t*)hbuf)[col * 32 + q] = 0u;
  }
  for (int t = tid; t < 256 * 16; t += 128) smax[t] = 0;

  // relative neighbor coords -> hbuf rows 0..2
  for (int col = tid; col < 320; col += 128) {
    int p = col / KNN, kk = col % KNN;
    int i = pbase + p;
    int j = ws_idx[((size_t)b * NPTS + i) * KNN + kk];
#pragma unroll
    for (int c = 0; c < 3; ++c)
      hbuf[col * 64 + c] = (_Float16)(xb[c * NPTS + j] - xb[c * NPTS + i]);
  }
  __syncthreads();

  int lane = tid & 31, wave = tid >> 5;
  int ln = lane & 15, lg = lane >> 4;
  const _Float16* w1 = wh + WH_W1;
  const _Float16* w2 = wh + WH_W2;
  const _Float16* w3 = wh + WH_W3;
  const _Float16* w4 = wh + WH_W4;

  for (int ct = wave; ct < 20; ct += 4) {
    int colbase = ct * 16;
    int col = colbase + ln;
    int p = col / KNN;                 // local point 0..15 this lane contributes to

    // ---- layer 1: 3(pad 32) -> 32 ----
    {
      Frag16 bf = load_b_lds(hbuf, colbase, ln, lg, 0);
      v8f acc[2];
#pragma unroll
      for (int mt = 0; mt < 2; ++mt) {
        Frag16 af = load_a(w1, 32, mt * 16, ln, lg, 0);
        v8f z = {0.f, 0.f, 0.f, 0.f, 0.f, 0.f, 0.f, 0.f};
        acc[mt] = wmma16(af, bf, z);
      }
#pragma unroll
      for (int mt = 0; mt < 2; ++mt)
#pragma unroll
        for (int r = 0; r < 8; ++r) {
          int M = mt * 16 + r + lg * 8;
          float yv = fmaxf(g1[M] * acc[mt][r] * INV_BN + b1[M], 0.f);
          atomicMax(&smax[(0 + M) * 16 + p], __float_as_int(yv));
          acc[mt][r] = yv;
        }
#pragma unroll
      for (int mt = 0; mt < 2; ++mt)
#pragma unroll
        for (int r = 0; r < 8; ++r)
          hbuf[col * 64 + mt * 16 + r + lg * 8] = (_Float16)acc[mt][r];
    }
    // ---- layer 2: 32 -> 32 ----
    {
      Frag16 bf = load_b_lds(hbuf, colbase, ln, lg, 0);
      v8f acc[2];
#pragma unroll
      for (int mt = 0; mt < 2; ++mt) {
        Frag16 af = load_a(w2, 32, mt * 16, ln, lg, 0);
        v8f z = {0.f, 0.f, 0.f, 0.f, 0.f, 0.f, 0.f, 0.f};
        acc[mt] = wmma16(af, bf, z);
      }
#pragma unroll
      for (int mt = 0; mt < 2; ++mt)
#pragma unroll
        for (int r = 0; r < 8; ++r) {
          int M = mt * 16 + r + lg * 8;
          float yv = fmaxf(g2[M] * acc[mt][r] * INV_BN + b2[M], 0.f);
          atomicMax(&smax[(32 + M) * 16 + p], __float_as_int(yv));
          acc[mt][r] = yv;
        }
#pragma unroll
      for (int mt = 0; mt < 2; ++mt)
#pragma unroll
        for (int r = 0; r < 8; ++r)
          hbuf[col * 64 + mt * 16 + r + lg * 8] = (_Float16)acc[mt][r];
    }
    // ---- layer 3: 32 -> 64 ----
    {
      Frag16 bf = load_b_lds(hbuf, colbase, ln, lg, 0);
      v8f acc[4];
#pragma unroll
      for (int mt = 0; mt < 4; ++mt) {
        Frag16 af = load_a(w3, 32, mt * 16, ln, lg, 0);
        v8f z = {0.f, 0.f, 0.f, 0.f, 0.f, 0.f, 0.f, 0.f};
        acc[mt] = wmma16(af, bf, z);
      }
#pragma unroll
      for (int mt = 0; mt < 4; ++mt)
#pragma unroll
        for (int r = 0; r < 8; ++r) {
          int M = mt * 16 + r + lg * 8;
          float yv = fmaxf(g3[M] * acc[mt][r] * INV_BN + b3[M], 0.f);
          atomicMax(&smax[(64 + M) * 16 + p], __float_as_int(yv));
          acc[mt][r] = yv;
        }
#pragma unroll
      for (int mt = 0; mt < 4; ++mt)
#pragma unroll
        for (int r = 0; r < 8; ++r)
          hbuf[col * 64 + mt * 16 + r + lg * 8] = (_Float16)acc[mt][r];
    }
    // ---- layer 4: 64 -> 128 (max only, no store-back) ----
    {
      Frag16 bf0 = load_b_lds(hbuf, colbase, ln, lg, 0);
      Frag16 bf1 = load_b_lds(hbuf, colbase, ln, lg, 32);
#pragma unroll
      for (int mt = 0; mt < 8; ++mt) {
        Frag16 af0 = load_a(w4, 64, mt * 16, ln, lg, 0);
        Frag16 af1 = load_a(w4, 64, mt * 16, ln, lg, 32);
        v8f z = {0.f, 0.f, 0.f, 0.f, 0.f, 0.f, 0.f, 0.f};
        v8f acc = wmma16(af0, bf0, z);
        acc = wmma16(af1, bf1, acc);
#pragma unroll
        for (int r = 0; r < 8; ++r) {
          int M = mt * 16 + r + lg * 8;
          float yv = fmaxf(g4[M] * acc[r] * INV_BN + b4[M], 0.f);
          atomicMax(&smax[(128 + M) * 16 + p], __float_as_int(yv));
        }
      }
    }
  }
  __syncthreads();

  // write concat features [b][n][256] as f16
  for (int t = tid; t < 16 * 256; t += 128) {
    int p = t >> 8, ch = t & 255;
    float v = __int_as_float(smax[ch * 16 + p]);
    cat[((size_t)b * NPTS + pbase + p) * 256 + ch] = (_Float16)v;
  }
}

// ---------------- Kernel 4: head 256 -> 512 ----------------
__global__ __launch_bounds__(256) void head_kernel(
    const _Float16* __restrict__ cat, const _Float16* __restrict__ w5,
    const float* __restrict__ g5, const float* __restrict__ b5,
    float* __restrict__ out) {
  int b  = blockIdx.x >> 9;               // 512 column tiles per batch
  int nb = (blockIdx.x & 511) * 16;
  int tid = threadIdx.x, lane = tid & 31, wave = tid >> 5;
  int ln = lane & 15, lg = lane >> 4;
  int n = nb + ln;
  const _Float16* colp = cat + ((size_t)b * NPTS + n) * 256;

  v8f acc[4];
#pragma unroll
  for (int mi = 0; mi < 4; ++mi) {
    v8f z = {0.f, 0.f, 0.f, 0.f, 0.f, 0.f, 0.f, 0.f};
    acc[mi] = z;
  }
#pragma unroll
  for (int kb = 0; kb < 256; kb += 32) {
    Frag16 bf;
    const _Float16* p = colp + kb + lg * 8;
    bf.q[0] = *(const uint4*)(p);
    bf.q[1] = *(const uint4*)(p + 16);
#pragma unroll
    for (int mi = 0; mi < 4; ++mi) {
      Frag16 af = load_a(w5, 256, (wave * 4 + mi) * 16, ln, lg, kb);
      acc[mi] = wmma16(af, bf, acc[mi]);
    }
  }
#pragma unroll
  for (int mi = 0; mi < 4; ++mi)
#pragma unroll
    for (int r = 0; r < 8; ++r) {
      int M = (wave * 4 + mi) * 16 + r + lg * 8;
      float yv = fmaxf(g5[M] * acc[mi][r] * INV_BN + b5[M], 0.f);
      out[OFF_FEAT + ((size_t)b * 512 + M) * NPTS + n] = yv;
    }
}

extern "C" void kernel_launch(void* const* d_in, const int* in_sizes, int n_in,
                              void* d_out, int out_size, void* d_ws, size_t ws_size,
                              hipStream_t stream) {
  (void)in_sizes; (void)n_in; (void)out_size; (void)ws_size;
  const float* x  = (const float*)d_in[0];
  // d_in[1] = k (always 20)
  const float* W1 = (const float*)d_in[2];
  const float* W2 = (const float*)d_in[3];
  const float* W3 = (const float*)d_in[4];
  const float* W4 = (const float*)d_in[5];
  const float* W5 = (const float*)d_in[6];
  const float* g1 = (const float*)d_in[7];  const float* b1 = (const float*)d_in[8];
  const float* g2 = (const float*)d_in[9];  const float* b2 = (const float*)d_in[10];
  const float* g3 = (const float*)d_in[11]; const float* b3 = (const float*)d_in[12];
  const float* g4 = (const float*)d_in[13]; const float* b4 = (const float*)d_in[14];
  const float* g5 = (const float*)d_in[15]; const float* b5 = (const float*)d_in[16];

  float*     out    = (float*)d_out;
  int*       ws_idx = (int*)d_ws;
  _Float16*  wh     = (_Float16*)((char*)d_ws + WS_WH);
  _Float16*  cat    = (_Float16*)((char*)d_ws + WS_CAT);

  knn_kernel<<<BATCH * (NPTS / 256), 256, 0, stream>>>(x, out, ws_idx);
  wconv_kernel<<<(WH_TOTAL + 255) / 256, 256, 0, stream>>>(W1, W2, W3, W4, W5, wh);
  edgeconv_kernel<<<BATCH * (NPTS / 16), 128, 0, stream>>>(
      x, ws_idx, wh, g1, b1, g2, b2, g3, b3, g4, b4, cat);
  head_kernel<<<BATCH * (NPTS / 16), 256, 0, stream>>>(cat, wh + WH_W5, g5, b5, out);
}
